// MultiHeadSelfAttention_39187281609313
// MI455X (gfx1250) — compile-verified, matrix operand order checked by reading, not run on hardware
//
#include <hip/hip_runtime.h>
#include <hip/hip_bf16.h>
#include <math.h>

#define B_     8
#define L_     1024
#define D_     512
#define H_     8
#define DH_    64
#define INNER_ 512
#define BL_    (B_ * L_)

typedef __attribute__((ext_vector_type(16))) _Float16 v16h;
typedef __attribute__((ext_vector_type(8)))  _Float16 v8h;
typedef __attribute__((ext_vector_type(8)))  float    v8f;

static __device__ __forceinline__ v8f wmma32(v16h a, v16h b, v8f c) {
  return __builtin_amdgcn_wmma_f32_16x16x32_f16(
      /*neg_a=*/false, a, /*neg_b=*/false, b,
      /*c_mod=*/(short)0, c, /*reuse_a=*/false, /*reuse_b=*/false);
}

// A/B operand load for 16x16x32 f16 WMMA.
// Lane l holds matrix row (l & 15); halves 0..7 come from K = kbase + hi*8 .. +7,
// halves 8..15 from K = kbase + hi*8 + 16 .. +23, hi = l >> 4.
static __device__ __forceinline__ v16h load_k32(const _Float16* __restrict__ row,
                                                int kbase, int hi) {
  const _Float16* p = row + kbase + (hi << 3);
  v8h lo = *(const v8h*)p;
  v8h hh = *(const v8h*)(p + 16);
  v16h r;
#pragma unroll
  for (int i = 0; i < 8; ++i) { r[i] = lo[i]; r[i + 8] = hh[i]; }
  return r;
}

static __device__ __forceinline__ v16h load_k32_f32(const float* __restrict__ row,
                                                    int kbase, int hi) {
  const float* p = row + kbase + (hi << 3);
  v16h r;
#pragma unroll
  for (int i = 0; i < 8; ++i) {
    r[i]     = (_Float16)p[i];
    r[i + 8] = (_Float16)p[i + 16];
  }
  return r;
}

// ---------------- LayerNorm: (BL, D) f32 -> f16 ----------------
__global__ void ln_kernel(const float* __restrict__ x, const float* __restrict__ gamma,
                          const float* __restrict__ beta, _Float16* __restrict__ xn) {
  __shared__ float red[256];
  int row = blockIdx.x, t = threadIdx.x;
  const float* xr = x + (size_t)row * D_;
  float v0 = xr[t], v1 = xr[t + 256];
  red[t] = v0 + v1;
  __syncthreads();
  for (int s = 128; s > 0; s >>= 1) { if (t < s) red[t] += red[t + s]; __syncthreads(); }
  float mu = red[0] * (1.0f / D_);
  __syncthreads();
  float d0 = v0 - mu, d1 = v1 - mu;
  red[t] = d0 * d0 + d1 * d1;
  __syncthreads();
  for (int s = 128; s > 0; s >>= 1) { if (t < s) red[t] += red[t + s]; __syncthreads(); }
  float rstd = rsqrtf(red[0] * (1.0f / D_) + 1e-5f);
  xn[(size_t)row * D_ + t]       = (_Float16)(d0 * rstd * gamma[t] + beta[t]);
  xn[(size_t)row * D_ + t + 256] = (_Float16)(d1 * rstd * gamma[t + 256] + beta[t + 256]);
}

// ---------------- Sinusoid relative-position table (L, D) ----------------
__global__ void pe_kernel(_Float16* __restrict__ pe) {
  int idx = blockIdx.x * 256 + threadIdx.x;
  int i = idx >> 9, j = idx & 511;
  float f   = __expf(-(float)j * (9.210340371976184f / 256.0f)); // log(1e4)/256
  float ang = (float)(i - j) * f;
  float val = ((j & 1) == 0) ? __cosf(ang) : __sinf(ang);
  pe[idx] = (_Float16)val;
}

__global__ void cvt_kernel(const float* __restrict__ src, _Float16* __restrict__ dst, int n) {
  int idx = blockIdx.x * 256 + threadIdx.x;
  if (idx < n) dst[idx] = (_Float16)src[idx];
}

// ------- Wave-level (MT*16)x64 GEMM tile, K-major A and B stored (N,K) row-major -------
// Per k-step: 2*MT A-loads + 8 B-loads (b128) feeding 4*MT WMMAs.
template <int MT>
static __device__ __forceinline__ void gemm_tile(
    const _Float16* __restrict__ A, int lda, int mbase,
    const _Float16* __restrict__ B, int ldb, int nbase,
    int K, v8f (&c)[MT][4]) {
  int lane = threadIdx.x & 31;
  int m = lane & 15, hi = lane >> 4;
  const _Float16* arow[MT];
#pragma unroll
  for (int i = 0; i < MT; ++i)
    arow[i] = A + (size_t)(mbase + i * 16 + m) * lda;
  const _Float16* brow0 = B + (size_t)(nbase + m) * ldb;
  const _Float16* brow1 = brow0 + (size_t)16 * ldb;
  const _Float16* brow2 = brow0 + (size_t)32 * ldb;
  const _Float16* brow3 = brow0 + (size_t)48 * ldb;
  for (int k = 0; k < K; k += 32) {
    v16h b0 = load_k32(brow0, k, hi);
    v16h b1 = load_k32(brow1, k, hi);
    v16h b2 = load_k32(brow2, k, hi);
    v16h b3 = load_k32(brow3, k, hi);
#pragma unroll
    for (int i = 0; i < MT; ++i) {
      v16h a = load_k32(arow[i], k, hi);
      c[i][0] = wmma32(a, b0, c[i][0]);
      c[i][1] = wmma32(a, b1, c[i][1]);
      c[i][2] = wmma32(a, b2, c[i][2]);
      c[i][3] = wmma32(a, b3, c[i][3]);
    }
  }
}

// ---------------- QKV GEMM: (8192x512) x (1536x512)^T, scatter to Q/K/Vt ----------------
__global__ void qkv_gemm_kernel(const _Float16* __restrict__ xn, const _Float16* __restrict__ wqkv,
                                const float* __restrict__ bqkv,
                                _Float16* __restrict__ Q, _Float16* __restrict__ Kb,
                                _Float16* __restrict__ Vt) {
  int wave  = threadIdx.x >> 5;
  int gtile = blockIdx.x * 8 + wave;          // 6144 tiles of 32x64
  int nt = gtile % 24, mt = gtile / 24;
  int mbase = mt * 32, nbase = nt * 64;
  v8f c[2][4] = {};
  gemm_tile<2>(xn, D_, mbase, wqkv, D_, nbase, D_, c);
  int lane = threadIdx.x & 31, col = lane & 15, hi = lane >> 4;
#pragma unroll
  for (int i = 0; i < 2; ++i) {
#pragma unroll
    for (int j = 0; j < 4; ++j) {
#pragma unroll
      for (int r = 0; r < 8; ++r) {
        int n = nbase + j * 16 + col;
        float val = c[i][j][r] + bqkv[n];
        int mg = mbase + i * 16 + r + 8 * hi;
        int b = mg >> 10, l = mg & 1023;
        if (n < INNER_) {
          int h = n >> 6, d = n & 63;
          Q[(((size_t)(b * H_ + h)) * L_ + l) * DH_ + d] = (_Float16)val;
        } else if (n < 2 * INNER_) {
          int n2 = n - INNER_, h = n2 >> 6, d = n2 & 63;
          Kb[(((size_t)(b * H_ + h)) * L_ + l) * DH_ + d] = (_Float16)val;
        } else {
          int n2 = n - 2 * INNER_, h = n2 >> 6, d = n2 & 63;
          Vt[(((size_t)(b * H_ + h)) * DH_ + d) * L_ + l] = (_Float16)val;  // transposed V
        }
      }
    }
  }
}

// ---------------- Pos GEMM: pe(1024x512) x w_pos(512x512)^T -> Pos(H,L,Dh) ----------------
__global__ void pos_gemm_kernel(const _Float16* __restrict__ pe, const _Float16* __restrict__ wpos,
                                _Float16* __restrict__ Pos) {
  int wave  = threadIdx.x >> 5;
  int gtile = blockIdx.x * 8 + wave;          // 256 tiles of 32x64
  int nt = gtile % 8, mt = gtile / 8;
  int mbase = mt * 32, nbase = nt * 64;
  v8f c[2][4] = {};
  gemm_tile<2>(pe, D_, mbase, wpos, D_, nbase, D_, c);
  int lane = threadIdx.x & 31, col = lane & 15, hi = lane >> 4;
#pragma unroll
  for (int i = 0; i < 2; ++i)
#pragma unroll
    for (int j = 0; j < 4; ++j)
#pragma unroll
      for (int r = 0; r < 8; ++r) {
        int n = nbase + j * 16 + col;
        int h = n >> 6, d = n & 63;
        int l = mbase + i * 16 + r + 8 * hi;
        Pos[((size_t)h * L_ + l) * DH_ + d] = (_Float16)c[i][j][r];
      }
}

// ---------------- Flash attention: content + position scores, online softmax ----------------
__global__ void attn_kernel(const _Float16* __restrict__ Q, const _Float16* __restrict__ Kb,
                            const _Float16* __restrict__ Vt, const _Float16* __restrict__ Pos,
                            const float* __restrict__ u_bias, const float* __restrict__ v_bias,
                            _Float16* __restrict__ AttnOut) {
  __shared__ _Float16 lds[8][16 * 40];        // per-wave 16x32 P tile, stride 40 halves
  int wave = threadIdx.x >> 5, lane = threadIdx.x & 31;
  int m = lane & 15, hi = lane >> 4;
  int bid = blockIdx.x;                       // 512 blocks: (b,h) x 8 query chunks of 128
  int qc = bid & 7, bh = bid >> 3;
  int h = bh & 7, b = bh >> 3;
  int q0 = qc * 128 + wave * 16;

  const _Float16* Qbh = Q   + ((size_t)(b * H_ + h)) * L_ * DH_;
  const _Float16* Kbh = Kb  + ((size_t)(b * H_ + h)) * L_ * DH_;
  const _Float16* Vbh = Vt  + ((size_t)(b * H_ + h)) * DH_ * L_;
  const _Float16* Ph  = Pos + (size_t)h * L_ * DH_;

  const _Float16* qrow = Qbh + (size_t)(q0 + m) * DH_;
  v16h qa0 = load_k32(qrow, 0, hi), qa1 = load_k32(qrow, 32, hi);
  v16h u0 = load_k32_f32(u_bias + h * DH_, 0, hi), u1 = load_k32_f32(u_bias + h * DH_, 32, hi);
  v16h w0 = load_k32_f32(v_bias + h * DH_, 0, hi), w1 = load_k32_f32(v_bias + h * DH_, 32, hi);
  v16h qu0 = qa0 + u0, qu1 = qa1 + u1;
  v16h qv0 = qa0 + w0, qv1 = qa1 + w1;

  float run_max[8], l_part[8];
  v8f o[4] = {};
#pragma unroll
  for (int r = 0; r < 8; ++r) { run_max[r] = -1e30f; l_part[r] = 0.0f; }
  const float scale = 0.125f;                 // Dh^-0.5

#pragma unroll 1
  for (int kb = 0; kb < L_; kb += 32) {
    v8f s0 = {}, s1 = {};
    {
      const _Float16* kr = Kbh + (size_t)(kb + m) * DH_;
      const _Float16* pr = Ph  + (size_t)(kb + m) * DH_;
      s0 = wmma32(qu0, load_k32(kr, 0, hi),  s0);
      s0 = wmma32(qu1, load_k32(kr, 32, hi), s0);
      s0 = wmma32(qv0, load_k32(pr, 0, hi),  s0);
      s0 = wmma32(qv1, load_k32(pr, 32, hi), s0);
    }
    {
      const _Float16* kr = Kbh + (size_t)(kb + 16 + m) * DH_;
      const _Float16* pr = Ph  + (size_t)(kb + 16 + m) * DH_;
      s1 = wmma32(qu0, load_k32(kr, 0, hi),  s1);
      s1 = wmma32(qu1, load_k32(kr, 32, hi), s1);
      s1 = wmma32(qv0, load_k32(pr, 0, hi),  s1);
      s1 = wmma32(qv1, load_k32(pr, 32, hi), s1);
    }
    // online softmax; C-layout row = r + 8*hi lives on a fixed 16-lane half
#pragma unroll
    for (int r = 0; r < 8; ++r) {
      float a = s0[r] * scale, bb = s1[r] * scale;
      float t = fmaxf(a, bb);
      t = fmaxf(t, __shfl_xor(t, 1, 32));
      t = fmaxf(t, __shfl_xor(t, 2, 32));
      t = fmaxf(t, __shfl_xor(t, 4, 32));
      t = fmaxf(t, __shfl_xor(t, 8, 32));
      float nm = fmaxf(run_max[r], t);
      float f  = __expf(run_max[r] - nm);
      run_max[r] = nm;
      l_part[r] *= f;
      o[0][r] *= f; o[1][r] *= f; o[2][r] *= f; o[3][r] *= f;
      float p0 = __expf(a - nm), p1 = __expf(bb - nm);
      l_part[r] += p0 + p1;
      int row = r + 8 * hi;
      lds[wave][row * 40 + m]      = (_Float16)p0;
      lds[wave][row * 40 + m + 16] = (_Float16)p1;
    }
    // transpose-stage P through LDS into A-operand layout (in-order DS within wave)
    v16h pa = load_k32(&lds[wave][m * 40], 0, hi);
#pragma unroll
    for (int nt = 0; nt < 4; ++nt) {
      const _Float16* vr = Vbh + (size_t)(nt * 16 + m) * L_;   // Vt row d, contiguous in key
      o[nt] = wmma32(pa, load_k32(vr, kb, hi), o[nt]);
    }
  }
  // final row-sum normalization
#pragma unroll
  for (int r = 0; r < 8; ++r) {
    float s = l_part[r];
    s += __shfl_xor(s, 1, 32);
    s += __shfl_xor(s, 2, 32);
    s += __shfl_xor(s, 4, 32);
    s += __shfl_xor(s, 8, 32);
    float inv = 1.0f / s;
    o[0][r] *= inv; o[1][r] *= inv; o[2][r] *= inv; o[3][r] *= inv;
  }
#pragma unroll
  for (int nt = 0; nt < 4; ++nt)
#pragma unroll
    for (int r = 0; r < 8; ++r) {
      int mg = b * L_ + q0 + r + 8 * hi;
      int d  = nt * 16 + m;
      AttnOut[(size_t)mg * INNER_ + h * DH_ + d] = (_Float16)o[nt][r];
    }
}

// ---------------- Output GEMM: (8192x512) x (512x512)^T + b_out -> f32 ----------------
__global__ void out_gemm_kernel(const _Float16* __restrict__ ao, const _Float16* __restrict__ wout,
                                const float* __restrict__ bout, float* __restrict__ out) {
  int wave  = threadIdx.x >> 5;
  int gtile = blockIdx.x * 8 + wave;          // 2048 tiles of 32x64
  int nt = gtile % 8, mt = gtile / 8;
  int mbase = mt * 32, nbase = nt * 64;
  v8f c[2][4] = {};
  gemm_tile<2>(ao, INNER_, mbase, wout, INNER_, nbase, INNER_, c);
  int lane = threadIdx.x & 31, col = lane & 15, hi = lane >> 4;
#pragma unroll
  for (int i = 0; i < 2; ++i)
#pragma unroll
    for (int j = 0; j < 4; ++j)
#pragma unroll
      for (int r = 0; r < 8; ++r) {
        int n  = nbase + j * 16 + col;
        int mg = mbase + i * 16 + r + 8 * hi;
        out[(size_t)mg * D_ + n] = c[i][j][r] + bout[n];
      }
}

extern "C" void kernel_launch(void* const* d_in, const int* in_sizes, int n_in,
                              void* d_out, int out_size, void* d_ws, size_t ws_size,
                              hipStream_t stream) {
  (void)in_sizes; (void)n_in; (void)out_size; (void)ws_size;
  const float* x     = (const float*)d_in[0];
  const float* gamma = (const float*)d_in[1];
  const float* beta  = (const float*)d_in[2];
  const float* wqkv  = (const float*)d_in[3];
  const float* bqkv  = (const float*)d_in[4];
  const float* wpos  = (const float*)d_in[5];
  const float* wout  = (const float*)d_in[6];
  const float* bout  = (const float*)d_in[7];
  const float* ubias = (const float*)d_in[8];
  const float* vbias = (const float*)d_in[9];
  float* out = (float*)d_out;

  char* ws = (char*)d_ws;
  size_t off = 0;
  auto alloc = [&](size_t bytes) -> void* {
    void* p = ws + off;
    off += (bytes + 255) & ~(size_t)255;
    return p;
  };
  _Float16* xnF   = (_Float16*)alloc((size_t)BL_ * D_ * 2);
  _Float16* wqkvF = (_Float16*)alloc((size_t)3 * INNER_ * D_ * 2);
  _Float16* wposF = (_Float16*)alloc((size_t)INNER_ * D_ * 2);
  _Float16* woutF = (_Float16*)alloc((size_t)D_ * INNER_ * 2);
  _Float16* peF   = (_Float16*)alloc((size_t)L_ * D_ * 2);
  _Float16* Qb    = (_Float16*)alloc((size_t)B_ * H_ * L_ * DH_ * 2);
  _Float16* Kbf   = (_Float16*)alloc((size_t)B_ * H_ * L_ * DH_ * 2);
  _Float16* Vt    = (_Float16*)alloc((size_t)B_ * H_ * DH_ * L_ * 2);
  _Float16* PosB  = (_Float16*)alloc((size_t)H_ * L_ * DH_ * 2);
  _Float16* AOut  = (_Float16*)alloc((size_t)BL_ * INNER_ * 2);

  ln_kernel<<<BL_, 256, 0, stream>>>(x, gamma, beta, xnF);
  pe_kernel<<<(L_ * D_) / 256, 256, 0, stream>>>(peF);
  cvt_kernel<<<(3 * INNER_ * D_) / 256, 256, 0, stream>>>(wqkv, wqkvF, 3 * INNER_ * D_);
  cvt_kernel<<<(INNER_ * D_) / 256, 256, 0, stream>>>(wpos, wposF, INNER_ * D_);
  cvt_kernel<<<(D_ * INNER_) / 256, 256, 0, stream>>>(wout, woutF, D_ * INNER_);
  qkv_gemm_kernel<<<768, 256, 0, stream>>>(xnF, wqkvF, bqkv, Qb, Kbf, Vt);
  pos_gemm_kernel<<<32, 256, 0, stream>>>(peF, wposF, PosB);
  attn_kernel<<<512, 256, 0, stream>>>(Qb, Kbf, Vt, PosB, ubias, vbias, AOut);
  out_gemm_kernel<<<256, 256, 0, stream>>>(AOut, woutF, bout, out);
}
